// InterChainAttention_75282186764817
// MI455X (gfx1250) — compile-verified, matrix operand order checked by reading, not run on hardware
//
#include <hip/hip_runtime.h>

typedef __bf16 bf16_t;
typedef __attribute__((ext_vector_type(16))) __bf16 v16bf;
typedef __attribute__((ext_vector_type(8)))  __bf16 v8bf;
typedef __attribute__((ext_vector_type(8)))  float  v8f;

#define WMMA_BF16(a, b, c) \
  __builtin_amdgcn_wmma_f32_16x16x32_bf16(false, (a), false, (b), (short)0, (c), false, false)

// 16-byte async global->LDS copy (per-lane). Tracked on ASYNCcnt.
__device__ __forceinline__ void async_copy16(const bf16_t* g, bf16_t* l) {
  unsigned lds = (unsigned)(size_t)l;  // low 32 bits of shared flat addr = LDS offset
  asm volatile("global_load_async_to_lds_b128 %0, %1, off"
               :: "v"(lds), "v"(g) : "memory");
}

// ---------------------------------------------------------------------------
// fp32 -> bf16 conversion
// ---------------------------------------------------------------------------
__global__ __launch_bounds__(256) void conv_bf16_kernel(const float* __restrict__ in,
                                                        bf16_t* __restrict__ out, int n) {
  int i = blockIdx.x * blockDim.x + threadIdx.x;
  if (i < n) out[i] = (bf16_t)in[i];
}

// ---------------------------------------------------------------------------
// C = A(bf16, MxK row-major) @ B(bf16, NxK row-major)^T + bias
// mode 0: fp32 row-major output [M,N]
// mode 1: bf16 output scattered to [B,H,S,hd] (m -> b,s ; n -> h,d)
// Block: 256 threads = 8 waves; tile 128x128; K-step 32; double-buffered LDS
// filled with global_load_async_to_lds_b128 (4 async/thread/buffer). Wave
// (wr 0..3, wc 0..1) owns a 32x64 output slab = 8 WMMA tiles -> 8 WMMAs per
// K-step per wave. Last K-step peeled so the hot loop is branch-free.
// ---------------------------------------------------------------------------
__global__ __launch_bounds__(256) void gemm_bf16_nt_kernel(
    const bf16_t* __restrict__ A, const bf16_t* __restrict__ Bw,
    const float* __restrict__ bias, float* __restrict__ outF, bf16_t* __restrict__ outH,
    int M, int N, int K, int mode, int S, int H, int hd) {
  __shared__ __align__(32) bf16_t Asm[2][128 * 32];
  __shared__ __align__(32) bf16_t Bsm[2][128 * 32];

  const int tid = threadIdx.x;
  const int wave = tid >> 5, lane = tid & 31;
  const int half = lane >> 4, lanelo = lane & 15;
  const int wr = wave & 3, wc = wave >> 2;
  const int m0 = blockIdx.y * 128, n0 = blockIdx.x * 128;

  // cooperative async fills: A and B are each 128x32 (32B/thread each)
  const int frow = tid >> 1, fcol = (tid & 1) * 16;

  v8f acc[2][4] = {};

  auto issue = [&](int buf, int kk) {
    const bf16_t* ga = &A[(size_t)(m0 + frow) * K + kk + fcol];
    bf16_t* la = &Asm[buf][frow * 32 + fcol];
    async_copy16(ga, la);
    async_copy16(ga + 8, la + 8);
    const bf16_t* gb = &Bw[(size_t)(n0 + frow) * K + kk + fcol];
    bf16_t* lb = &Bsm[buf][frow * 32 + fcol];
    async_copy16(gb, lb);
    async_copy16(gb + 8, lb + 8);
  };

  auto compute = [&](int cur) {
    v16bf a0 = *(const v16bf*)&Asm[cur][(wr * 32 + lanelo) * 32 + half * 16];
    v16bf a1 = *(const v16bf*)&Asm[cur][(wr * 32 + 16 + lanelo) * 32 + half * 16];
#pragma unroll
    for (int t = 0; t < 4; ++t) {
      v16bf bt = *(const v16bf*)&Bsm[cur][(wc * 64 + t * 16 + lanelo) * 32 + half * 16];
      acc[0][t] = WMMA_BF16(a0, bt, acc[0][t]);
      acc[1][t] = WMMA_BF16(a1, bt, acc[1][t]);
    }
  };

  const int nsteps = K / 32;
  issue(0, 0);

  for (int s = 0; s < nsteps - 1; ++s) {
    const int cur = s & 1;
    __syncthreads();  // previous readers of buf cur^1 done -> safe to refill
    issue(cur ^ 1, (s + 1) * 32);
    asm volatile("s_wait_asynccnt 4" ::: "memory");  // cur landed, next in flight
    __syncthreads();  // publish cur buffer to all waves
    compute(cur);
  }
  __syncthreads();
  asm volatile("s_wait_asynccnt 0" ::: "memory");
  __syncthreads();
  compute((nsteps - 1) & 1);

#pragma unroll
  for (int i = 0; i < 2; ++i) {
#pragma unroll
    for (int t = 0; t < 4; ++t) {
      int gn = n0 + wc * 64 + t * 16 + lanelo;
      float bb = bias[gn];
#pragma unroll
      for (int j = 0; j < 8; ++j) {
        int gm = m0 + wr * 32 + i * 16 + half * 8 + j;
        float v = acc[i][t][j] + bb;
        if (mode == 0) {
          outF[(size_t)gm * N + gn] = v;
        } else {
          int b2 = gm / S, ss = gm % S;
          int h2 = gn / hd, d2 = gn % hd;
          outH[(((size_t)(b2 * H + h2)) * S + ss) * hd + d2] = (bf16_t)v;
        }
      }
    }
  }
}

// ---------------------------------------------------------------------------
// Flash attention with chain/distance bias.
// Grid: B*H*(S/128) blocks; 256 threads = 8 waves; each wave owns 16 query rows.
// Key tiles of 32 shared across the block: K via async-to-LDS, V via VGPR
// transpose into VsmT.
// ---------------------------------------------------------------------------
__global__ __launch_bounds__(256) void attn_kernel(
    const bf16_t* __restrict__ Qb, const bf16_t* __restrict__ Kb,
    const bf16_t* __restrict__ Vb, const int* __restrict__ chain,
    const float* __restrict__ dist, const float* __restrict__ chain_bias,
    const float* __restrict__ dist_w, const float* __restrict__ dist_b,
    bf16_t* __restrict__ AO) {
  const int S = 2048, D = 1024, H = 16, HD = 64;
  __shared__ __align__(32) bf16_t Ksm[32 * 64];   // [key][dim]
  __shared__ __align__(32) bf16_t VsmT[64 * 32];  // [dim][key]
  __shared__ __align__(32) bf16_t Psm[8 * 16 * 32];

  const int tid = threadIdx.x;
  const int wave = tid >> 5, lane = tid & 31;
  const int half = lane >> 4, lanelo = lane & 15;
  const int bh = blockIdx.x >> 4;   // b*H + h
  const int qblk = blockIdx.x & 15;
  const int b = bh >> 4, h = bh & 15;
  const int q0 = qblk * 128 + wave * 16;

  const bf16_t* Qbh = Qb + (size_t)bh * S * HD;
  const bf16_t* Kbh = Kb + (size_t)bh * S * HD;
  const bf16_t* Vbh = Vb + (size_t)bh * S * HD;
  bf16_t* pw = Psm + wave * 16 * 32;

  const float cb = chain_bias[h], dw = dist_w[h], dbv = dist_b[h];

  // Q A-operands for K-dim = 64 (two 32-chunks), held for the whole loop.
  v16bf aq0 = *(const v16bf*)(Qbh + (size_t)(q0 + lanelo) * HD + 0 * 32 + half * 16);
  v16bf aq1 = *(const v16bf*)(Qbh + (size_t)(q0 + lanelo) * HD + 1 * 32 + half * 16);

  int cq[8];
#pragma unroll
  for (int j = 0; j < 8; ++j) cq[j] = chain[q0 + half * 8 + j];

  v8f co0 = {}, co1 = {}, co2 = {}, co3 = {};
  float mrun[8], lrun[8];
#pragma unroll
  for (int j = 0; j < 8; ++j) { mrun[j] = -3.0e38f; lrun[j] = 0.f; }

  const int crow = tid >> 3, cseg = tid & 7;  // coop loads: 32 rows x 8 segs x 8 elems

  for (int kt = 0; kt < 64; ++kt) {
    const int k0 = kt * 32;
    __syncthreads();  // previous iteration's readers done
    {
      // K tile: async global->LDS, row-major (exactly the B-operand layout)
      async_copy16(Kbh + (size_t)(k0 + crow) * HD + cseg * 8,
                   &Ksm[crow * 64 + cseg * 8]);
      // V tile: through VGPRs with transpose into VsmT
      v8bf vv = *(const v8bf*)(Vbh + (size_t)(k0 + crow) * HD + cseg * 8);
#pragma unroll
      for (int i = 0; i < 8; ++i) VsmT[(cseg * 8 + i) * 32 + crow] = vv[i];
      if (kt + 1 < 64) {  // prefetch next tiles into cache
        __builtin_prefetch(Kbh + (size_t)(k0 + 32 + crow) * HD + cseg * 8, 0, 1);
        __builtin_prefetch(Vbh + (size_t)(k0 + 32 + crow) * HD + cseg * 8, 0, 1);
      }
    }
    asm volatile("s_wait_asynccnt 0" ::: "memory");
    __syncthreads();  // K/V tiles visible to all waves

    // scores: 16 q-rows x 32 keys = two WMMA C tiles, K-dim 64 = 2 chunks each.
    v8f sc0 = {}, sc1 = {};
    {
      v16bf bk;
      bk = *(const v16bf*)&Ksm[(0 * 16 + lanelo) * 64 + 0 * 32 + half * 16];
      sc0 = WMMA_BF16(aq0, bk, sc0);
      bk = *(const v16bf*)&Ksm[(0 * 16 + lanelo) * 64 + 1 * 32 + half * 16];
      sc0 = WMMA_BF16(aq1, bk, sc0);
      bk = *(const v16bf*)&Ksm[(1 * 16 + lanelo) * 64 + 0 * 32 + half * 16];
      sc1 = WMMA_BF16(aq0, bk, sc1);
      bk = *(const v16bf*)&Ksm[(1 * 16 + lanelo) * 64 + 1 * 32 + half * 16];
      sc1 = WMMA_BF16(aq1, bk, sc1);
    }

    const int ck0 = chain[k0 + lanelo];
    const int ck1 = chain[k0 + 16 + lanelo];

#pragma unroll
    for (int j = 0; j < 8; ++j) {
      const int qrow = q0 + half * 8 + j;
      const float* drow = dist + (size_t)qrow * S + k0 + lanelo;
      float s0 = sc0[j] * 0.125f + ((cq[j] != ck0) ? cb : 0.f) + drow[0] * dw + dbv;
      float s1 = sc1[j] * 0.125f + ((cq[j] != ck1) ? cb : 0.f) + drow[16] * dw + dbv;

      // row max across the 16 lanes holding this row's keys
      float mn = fmaxf(s0, s1);
      mn = fmaxf(mn, __shfl_xor(mn, 8, 16));
      mn = fmaxf(mn, __shfl_xor(mn, 4, 16));
      mn = fmaxf(mn, __shfl_xor(mn, 2, 16));
      mn = fmaxf(mn, __shfl_xor(mn, 1, 16));
      mn = fmaxf(mn, mrun[j]);
      float corr = __expf(mrun[j] - mn);
      mrun[j] = mn;

      float p0 = __expf(s0 - mn), p1 = __expf(s1 - mn);
      float rs = p0 + p1;
      rs += __shfl_xor(rs, 8, 16);
      rs += __shfl_xor(rs, 4, 16);
      rs += __shfl_xor(rs, 2, 16);
      rs += __shfl_xor(rs, 1, 16);
      lrun[j] = lrun[j] * corr + rs;

      co0[j] *= corr; co1[j] *= corr; co2[j] *= corr; co3[j] *= corr;

      // stage P (C layout -> A layout transpose through per-wave LDS)
      pw[(half * 8 + j) * 32 + lanelo] = (bf16_t)p0;
      pw[(half * 8 + j) * 32 + 16 + lanelo] = (bf16_t)p1;
    }

    asm volatile("s_wait_dscnt 0" ::: "memory");

    v16bf pa = *(const v16bf*)&pw[lanelo * 32 + half * 16];
    v16bf bv;
    bv = *(const v16bf*)&VsmT[(0 * 16 + lanelo) * 32 + half * 16]; co0 = WMMA_BF16(pa, bv, co0);
    bv = *(const v16bf*)&VsmT[(1 * 16 + lanelo) * 32 + half * 16]; co1 = WMMA_BF16(pa, bv, co1);
    bv = *(const v16bf*)&VsmT[(2 * 16 + lanelo) * 32 + half * 16]; co2 = WMMA_BF16(pa, bv, co2);
    bv = *(const v16bf*)&VsmT[(3 * 16 + lanelo) * 32 + half * 16]; co3 = WMMA_BF16(pa, bv, co3);
  }

  // normalize and write bf16 attention output in [B,S,D] (head-concat) layout
#pragma unroll
  for (int j = 0; j < 8; ++j) {
    float inv = 1.0f / lrun[j];
    int row = q0 + half * 8 + j;
    bf16_t* orow = AO + ((size_t)(b * S + row)) * D + h * HD;
    orow[0 * 16 + lanelo] = (bf16_t)(co0[j] * inv);
    orow[1 * 16 + lanelo] = (bf16_t)(co1[j] * inv);
    orow[2 * 16 + lanelo] = (bf16_t)(co2[j] * inv);
    orow[3 * 16 + lanelo] = (bf16_t)(co3[j] * inv);
  }
}

// ---------------------------------------------------------------------------
extern "C" void kernel_launch(void* const* d_in, const int* in_sizes, int n_in,
                              void* d_out, int out_size, void* d_ws, size_t ws_size,
                              hipStream_t stream) {
  (void)in_sizes; (void)n_in; (void)out_size; (void)ws_size;
  const int B = 2, S = 2048, D = 1024, H = 16, HD = 64;
  const int M = B * S;

  const float* x     = (const float*)d_in[0];
  const int*   chain = (const int*)d_in[1];
  const float* dist  = (const float*)d_in[2];
  const float* Wq = (const float*)d_in[3];  const float* bq = (const float*)d_in[4];
  const float* Wk = (const float*)d_in[5];  const float* bk = (const float*)d_in[6];
  const float* Wv = (const float*)d_in[7];  const float* bv = (const float*)d_in[8];
  const float* Wo = (const float*)d_in[9];  const float* bo = (const float*)d_in[10];
  const float* chb = (const float*)d_in[11];
  const float* dw  = (const float*)d_in[12];
  const float* dbv = (const float*)d_in[13];
  float* out = (float*)d_out;

  const size_t MB = 1024 * 1024;
  char* ws = (char*)d_ws;
  bf16_t* xb  = (bf16_t*)(ws + 0 * MB);   // 8 MB
  bf16_t* wqb = (bf16_t*)(ws + 8 * MB);   // 2 MB
  bf16_t* wkb = (bf16_t*)(ws + 10 * MB);  // 2 MB
  bf16_t* wvb = (bf16_t*)(ws + 12 * MB);  // 2 MB
  bf16_t* wob = (bf16_t*)(ws + 14 * MB);  // 2 MB
  bf16_t* qb  = (bf16_t*)(ws + 16 * MB);  // 8 MB  [B,H,S,hd]
  bf16_t* kb  = (bf16_t*)(ws + 24 * MB);  // 8 MB
  bf16_t* vb  = (bf16_t*)(ws + 32 * MB);  // 8 MB
  bf16_t* ao  = (bf16_t*)(ws + 40 * MB);  // 8 MB  [B,S,D]

  const int nx = M * D, nw = D * D;
  conv_bf16_kernel<<<(nx + 255) / 256, 256, 0, stream>>>(x, xb, nx);
  conv_bf16_kernel<<<(nw + 255) / 256, 256, 0, stream>>>(Wq, wqb, nw);
  conv_bf16_kernel<<<(nw + 255) / 256, 256, 0, stream>>>(Wk, wkb, nw);
  conv_bf16_kernel<<<(nw + 255) / 256, 256, 0, stream>>>(Wv, wvb, nw);
  conv_bf16_kernel<<<(nw + 255) / 256, 256, 0, stream>>>(Wo, wob, nw);

  dim3 ggrid(D / 128, M / 128);
  gemm_bf16_nt_kernel<<<ggrid, 256, 0, stream>>>(xb, wqb, bq, nullptr, qb,
                                                 M, D, D, 1, S, H, HD);
  gemm_bf16_nt_kernel<<<ggrid, 256, 0, stream>>>(xb, wkb, bk, nullptr, kb,
                                                 M, D, D, 1, S, H, HD);
  gemm_bf16_nt_kernel<<<ggrid, 256, 0, stream>>>(xb, wvb, bv, nullptr, vb,
                                                 M, D, D, 1, S, H, HD);

  attn_kernel<<<B * H * (S / 128), 256, 0, stream>>>(qb, kb, vb, chain, dist,
                                                     chb, dw, dbv, ao);

  gemm_bf16_nt_kernel<<<ggrid, 256, 0, stream>>>(ao, wob, bo, out, nullptr,
                                                 M, D, D, 0, S, H, HD);
}
